// NodeEdgeAttBlock_32573031973557
// MI455X (gfx1250) — matrix operand order repaired and checked by reading, not compiled, not run on hardware
//
#include <hip/hip_runtime.h>
#include <hip/hip_bf16.h>
#include <math.h>

// ---------------------------------------------------------------------------
// NodeEdgeAttBlock for gfx1250 (MI455X, wave32, WMMA).
// Shapes: BS=2, N=256, NH_N=256, NH_E=128, H=8, F=32.
// Strategy:
//   prep    : transpose weights to bf16 W^T[n][k] (WMMA B-operand friendly)
//   qkv     : q/k/v = x@W + b           (bf16 WMMA, fp32 accum)
//   edge    : per (b,i) workgroup:
//               phase1: y = q*k*s*(a@Wmul+bmul+1) + a@Wadd+badd -> LDS (bf16)
//               phase2: ahat = y@Wao + bao -> d_out          (WMMA from LDS)
//               phase3: per-channel softmax over keys + weighted-v -> ws
//   xout    : xhat = wv@Wxo + bxo -> d_out                   (bf16 WMMA)
// Note: edge_kernel uses 128KB dynamic LDS (within 320KB/WGP on CDNA5).
// ---------------------------------------------------------------------------

typedef __attribute__((ext_vector_type(16))) __bf16 v16bf;
typedef __attribute__((ext_vector_type(8)))  float  v8f;
typedef __attribute__((ext_vector_type(4)))  float  f4;
typedef __attribute__((ext_vector_type(4)))  unsigned int u4;

union V16 { v16bf v; u4 u[2]; };

__device__ __forceinline__ v8f wmma_bf16(v16bf a, v16bf b, v8f c) {
  // D(f32,16x16) = A(bf16,16x32) * B(bf16,32x16) + C
  return __builtin_amdgcn_wmma_f32_16x16x32_bf16(
      /*neg_a=*/false, a, /*neg_b=*/false, b,
      /*c_mod=*/(short)0, c, /*reuse_a=*/false, /*reuse_b=*/false);
}

// Build a 16x32 bf16 A-fragment for this lane from an fp32 row-major source.
// ISA layout: elements 0..7 <- K+0..7, elements 8..15 <- K+16..23 (caller
// already folded the lane>=16 "+8" K offset into p).
__device__ __forceinline__ v16bf pack_f32_frag(const float* p) {
  f4 a = *(const f4*)(p);
  f4 b = *(const f4*)(p + 4);
  f4 c = *(const f4*)(p + 16);
  f4 d = *(const f4*)(p + 20);
  v16bf v;
  v[0]  = (__bf16)a[0];  v[1]  = (__bf16)a[1];  v[2]  = (__bf16)a[2];  v[3]  = (__bf16)a[3];
  v[4]  = (__bf16)b[0];  v[5]  = (__bf16)b[1];  v[6]  = (__bf16)b[2];  v[7]  = (__bf16)b[3];
  v[8]  = (__bf16)c[0];  v[9]  = (__bf16)c[1];  v[10] = (__bf16)c[2];  v[11] = (__bf16)c[3];
  v[12] = (__bf16)d[0];  v[13] = (__bf16)d[1];  v[14] = (__bf16)d[2];  v[15] = (__bf16)d[3];
  return v;
}

// ------------------------------ prep ---------------------------------------
// dst[n*K + k] = (bf16) src[k*N + n]   (src is K x N row-major)
__global__ void transpose_to_bf16(const float* __restrict__ src,
                                  __bf16* __restrict__ dst, int K, int N) {
  int idx = blockIdx.x * blockDim.x + threadIdx.x;
  if (idx >= K * N) return;
  int n = idx / K;
  int k = idx - n * K;
  dst[idx] = (__bf16)src[k * N + n];
}

// ------------------------------ qkv ----------------------------------------
// q/k/v (512x256) = x(512x256) @ W(256x256) + b.  32 blocks x 256 thr.
__global__ __launch_bounds__(256) void qkv_kernel(
    const float* __restrict__ x,
    const __bf16* __restrict__ WqT, const __bf16* __restrict__ WkT,
    const __bf16* __restrict__ WvT,
    const float* __restrict__ bq, const float* __restrict__ bk,
    const float* __restrict__ bv,
    float* __restrict__ qo, float* __restrict__ ko, float* __restrict__ vo) {
  const int mt   = blockIdx.x;            // M strip (16 rows of x)
  const int lane = threadIdx.x & 31;
  const int wave = threadIdx.x >> 5;
  const int mrow = lane & 15;
  const int hsel = lane >> 4;

  // Hold the whole K=256 A-strip (x rows) as 8 bf16 fragments.
  V16 af[8];
  const float* abase = x + (size_t)(mt * 16 + mrow) * 256 + hsel * 8;
#pragma unroll
  for (int ks = 0; ks < 8; ++ks) af[ks].v = pack_f32_frag(abase + ks * 32);

  // 48 output tiles (3 mats x 16 N-tiles) over 8 waves -> 6 tiles/wave.
  for (int ii = 0; ii < 6; ++ii) {
    const int t   = wave * 6 + ii;
    const int mat = t >> 4;
    const int nt  = t & 15;
    const int n   = nt * 16 + mrow;

    const __bf16* wt;
    const float*  bp;
    float*        op;
    if (mat == 0)      { wt = WqT; bp = bq; op = qo; }
    else if (mat == 1) { wt = WkT; bp = bk; op = ko; }
    else               { wt = WvT; bp = bv; op = vo; }

    const __bf16* bb = wt + (size_t)n * 256 + hsel * 8;
    v8f acc = {};
#pragma unroll
    for (int ks = 0; ks < 8; ++ks) {
      V16 bfg;
      bfg.u[0] = *(const u4*)(bb + ks * 32);
      bfg.u[1] = *(const u4*)(bb + ks * 32 + 16);
      acc = wmma_bf16(af[ks].v, bfg.v, acc);
    }
    const float bias = bp[n];
#pragma unroll
    for (int r = 0; r < 8; ++r) {
      const int mg = mt * 16 + r + 8 * hsel;
      op[(size_t)mg * 256 + n] = acc[r] + bias;
    }
  }
}

// ------------------------------ edge ---------------------------------------
// One workgroup per (b,i).  LDS holds y (256 keys x 256 channels) in bf16.
__global__ __launch_bounds__(256) void edge_kernel(
    const float* __restrict__ a,
    const __bf16* __restrict__ WmulT, const __bf16* __restrict__ WaddT,
    const __bf16* __restrict__ WaoT,
    const float* __restrict__ bmul, const float* __restrict__ badd,
    const float* __restrict__ bao,
    const float* __restrict__ qws, const float* __restrict__ kws,
    const float* __restrict__ vws,
    float* __restrict__ wvout, float* __restrict__ ahat) {
  extern __shared__ __bf16 ybuf[];        // 256*256 bf16 = 128 KB
  const int bi   = blockIdx.x;            // b*256 + i
  const int b    = bi >> 8;
  const int lane = threadIdx.x & 31;
  const int wave = threadIdx.x >> 5;
  const int mrow = lane & 15;
  const int hsel = lane >> 4;
  const float inv_sqrt = 0.17677669529663687f;  // 1/sqrt(32)

  // ---- Phase 1: y[j][c] into LDS ----
  for (int s = 0; s < 2; ++s) {
    const int jt = wave * 2 + s;          // key strip 0..15
    // a strip (16 keys x 128 edge-ch) held in registers as 4 bf16 A-frags
    const float* abase =
        a + ((size_t)bi * 256 + (jt * 16 + mrow)) * 128 + hsel * 8;
    V16 af[4];
#pragma unroll
    for (int ks = 0; ks < 4; ++ks) af[ks].v = pack_f32_frag(abase + ks * 32);

    for (int nt = 0; nt < 16; ++nt) {
      const int nn = nt * 16 + mrow;      // channel c for this lane
      const __bf16* bm = WmulT + (size_t)nn * 128 + hsel * 8;
      const __bf16* bd = WaddT + (size_t)nn * 128 + hsel * 8;
      v8f acc1 = {};
      v8f acc2 = {};
#pragma unroll
      for (int ks = 0; ks < 4; ++ks) {
        V16 b1, b2;
        b1.u[0] = *(const u4*)(bm + ks * 32);
        b1.u[1] = *(const u4*)(bm + ks * 32 + 16);
        b2.u[0] = *(const u4*)(bd + ks * 32);
        b2.u[1] = *(const u4*)(bd + ks * 32 + 16);
        acc1 = wmma_bf16(af[ks].v, b1.v, acc1);
        acc2 = wmma_bf16(af[ks].v, b2.v, acc2);
      }
      const float qv  = qws[(size_t)bi * 256 + nn];
      const float bmc = bmul[nn];
      const float bdc = badd[nn];
#pragma unroll
      for (int r = 0; r < 8; ++r) {
        const int j  = jt * 16 + r + 8 * hsel;
        const float kv = kws[((size_t)(b * 256 + j)) * 256 + nn];
        const float yv =
            qv * kv * inv_sqrt * (acc1[r] + bmc + 1.0f) + acc2[r] + bdc;
        ybuf[j * 256 + nn] = (__bf16)yv;
      }
    }
  }
  __syncthreads();

  // ---- Phase 2: ahat[j][e] = y @ Wao + bao ----
  for (int s = 0; s < 2; ++s) {
    const int jt = wave * 2 + s;
    V16 yf[8];
    const __bf16* ybase = ybuf + (jt * 16 + mrow) * 256 + hsel * 8;
#pragma unroll
    for (int ks = 0; ks < 8; ++ks) {
      yf[ks].u[0] = *(const u4*)(ybase + ks * 32);        // K+0..7
      yf[ks].u[1] = *(const u4*)(ybase + ks * 32 + 16);   // K+16..23
    }
    for (int et = 0; et < 8; ++et) {
      const int e = et * 16 + mrow;
      const __bf16* bb = WaoT + (size_t)e * 256 + hsel * 8;
      v8f acc = {};
#pragma unroll
      for (int ks = 0; ks < 8; ++ks) {
        V16 bfg;
        bfg.u[0] = *(const u4*)(bb + ks * 32);
        bfg.u[1] = *(const u4*)(bb + ks * 32 + 16);
        acc = wmma_bf16(yf[ks].v, bfg.v, acc);
      }
      const float be = bao[e];
#pragma unroll
      for (int r = 0; r < 8; ++r) {
        const int j = jt * 16 + r + 8 * hsel;
        ahat[((size_t)bi * 256 + j) * 128 + e] = acc[r] + be;
      }
    }
  }

  // ---- Phase 3: per-channel softmax over keys + weighted v ----
  {
    const int c = threadIdx.x;            // channel 0..255
    float mx = -1e30f;
    for (int j = 0; j < 256; ++j)
      mx = fmaxf(mx, (float)ybuf[j * 256 + c]);
    float ssum = 0.0f, acc = 0.0f;
    const float* vcol = vws + (size_t)b * 65536 + c;
    for (int j = 0; j < 256; ++j) {
      const float ev = __expf((float)ybuf[j * 256 + c] - mx);
      ssum += ev;
      acc  += ev * vcol[(size_t)j * 256];
    }
    wvout[(size_t)bi * 256 + c] = acc / ssum;
  }
}

// ------------------------------ xout ---------------------------------------
// xhat (512x256) = wv(512x256) @ Wxo(256x256) + bxo.  32 blocks x 256 thr.
__global__ __launch_bounds__(256) void xout_kernel(
    const float* __restrict__ wv, const __bf16* __restrict__ WxoT,
    const float* __restrict__ bxo, float* __restrict__ out) {
  const int mt   = blockIdx.x;
  const int lane = threadIdx.x & 31;
  const int wave = threadIdx.x >> 5;
  const int mrow = lane & 15;
  const int hsel = lane >> 4;

  V16 af[8];
  const float* abase = wv + (size_t)(mt * 16 + mrow) * 256 + hsel * 8;
#pragma unroll
  for (int ks = 0; ks < 8; ++ks) af[ks].v = pack_f32_frag(abase + ks * 32);

  for (int s = 0; s < 2; ++s) {
    const int nt = wave * 2 + s;
    const int n  = nt * 16 + mrow;
    const __bf16* bb = WxoT + (size_t)n * 256 + hsel * 8;
    v8f acc = {};
#pragma unroll
    for (int ks = 0; ks < 8; ++ks) {
      V16 bfg;
      bfg.u[0] = *(const u4*)(bb + ks * 32);
      bfg.u[1] = *(const u4*)(bb + ks * 32 + 16);
      acc = wmma_bf16(af[ks].v, bfg.v, acc);
    }
    const float bias = bxo[n];
#pragma unroll
    for (int r = 0; r < 8; ++r) {
      const int mg = mt * 16 + r + 8 * hsel;
      out[(size_t)mg * 256 + n] = acc[r] + bias;
    }
  }
}

// ------------------------------ host ---------------------------------------
// Workspace layout (bytes):
static const size_t WQT_OFF  = 0;               // 256x256 bf16
static const size_t WKT_OFF  = 131072;          // 256x256 bf16
static const size_t WVT_OFF  = 262144;          // 256x256 bf16
static const size_t WXOT_OFF = 393216;          // 256x256 bf16
static const size_t WMULT_OFF= 524288;          // 256x128 bf16
static const size_t WADDT_OFF= 589824;          // 256x128 bf16
static const size_t WAOT_OFF = 655360;          // 128x256 bf16
static const size_t Q_OFF    = 720896;          // 512x256 f32
static const size_t K_OFF    = Q_OFF + 524288;
static const size_t V_OFF    = K_OFF + 524288;
static const size_t WV_OFF   = V_OFF + 524288;  // 512x256 f32 (weighted v)

extern "C" void kernel_launch(void* const* d_in, const int* in_sizes, int n_in,
                              void* d_out, int out_size, void* d_ws,
                              size_t ws_size, hipStream_t stream) {
  (void)in_sizes; (void)n_in; (void)out_size; (void)ws_size;
  const float* x    = (const float*)d_in[0];
  const float* a    = (const float*)d_in[1];
  const float* Wq   = (const float*)d_in[2];
  const float* bq   = (const float*)d_in[3];
  const float* Wk   = (const float*)d_in[4];
  const float* bk   = (const float*)d_in[5];
  const float* Wv   = (const float*)d_in[6];
  const float* bv   = (const float*)d_in[7];
  const float* Wmul = (const float*)d_in[8];
  const float* bmul = (const float*)d_in[9];
  const float* Wadd = (const float*)d_in[10];
  const float* badd = (const float*)d_in[11];
  const float* Wxo  = (const float*)d_in[12];
  const float* bxo  = (const float*)d_in[13];
  const float* Wao  = (const float*)d_in[14];
  const float* bao  = (const float*)d_in[15];

  char* ws = (char*)d_ws;
  __bf16* WqT   = (__bf16*)(ws + WQT_OFF);
  __bf16* WkT   = (__bf16*)(ws + WKT_OFF);
  __bf16* WvT   = (__bf16*)(ws + WVT_OFF);
  __bf16* WxoT  = (__bf16*)(ws + WXOT_OFF);
  __bf16* WmulT = (__bf16*)(ws + WMULT_OFF);
  __bf16* WaddT = (__bf16*)(ws + WADDT_OFF);
  __bf16* WaoT  = (__bf16*)(ws + WAOT_OFF);
  float*  qo    = (float*)(ws + Q_OFF);
  float*  ko    = (float*)(ws + K_OFF);
  float*  vo    = (float*)(ws + V_OFF);
  float*  wvb   = (float*)(ws + WV_OFF);

  float* out  = (float*)d_out;            // xhat: 512*256
  float* ahat = out + 512 * 256;          // ahat: 512*256*128

  // prep: weight transposes to bf16
  transpose_to_bf16<<<256, 256, 0, stream>>>(Wq,   WqT,   256, 256);
  transpose_to_bf16<<<256, 256, 0, stream>>>(Wk,   WkT,   256, 256);
  transpose_to_bf16<<<256, 256, 0, stream>>>(Wv,   WvT,   256, 256);
  transpose_to_bf16<<<256, 256, 0, stream>>>(Wxo,  WxoT,  256, 256);
  transpose_to_bf16<<<128, 256, 0, stream>>>(Wmul, WmulT, 128, 256);
  transpose_to_bf16<<<128, 256, 0, stream>>>(Wadd, WaddT, 128, 256);
  transpose_to_bf16<<<128, 256, 0, stream>>>(Wao,  WaoT,  256, 128);

  // q/k/v projections
  qkv_kernel<<<32, 256, 0, stream>>>(x, WqT, WkT, WvT, bq, bk, bv, qo, ko, vo);

  // fused edge gating + ahat + softmax + weighted-v  (128 KB dynamic LDS)
  edge_kernel<<<512, 256, 131072, stream>>>(a, WmulT, WaddT, WaoT, bmul, badd,
                                            bao, qo, ko, vo, wvb, ahat);

  // xhat projection
  xout_kernel<<<32, 256, 0, stream>>>(wvb, WxoT, bxo, out);
}
